// StateCritic_66907000537303
// MI455X (gfx1250) — matrix operand order repaired and checked by reading, not compiled, not run on hardware
//
#include <hip/hip_runtime.h>
#include <hip/hip_bf16.h>

typedef __attribute__((ext_vector_type(16))) _Float16 v16h;
typedef __attribute__((ext_vector_type(8)))  float    v8f;

#define Hd 128
#define SEQ 1000
#define BATCH 256

// ---------------------------------------------------------------------------
// Kernel 1: fold W (H,3H) + encoder weights into per-h coefficient vectors.
//   pre[b,h,s] = a0[h]*x0 + a1[h]*x1 + bd[h]*y + cbase[h] + [q0[h]*px0 + q1[h]*px1 + qb[h]]
// ws layout (floats): a0[128] a1[128] bd[128] cbase[128] q0[128] q1[128] qb[128]
// ---------------------------------------------------------------------------
__global__ __launch_bounds__(128)
void fold_kernel(const float* __restrict__ W,
                 const float* __restrict__ esw, const float* __restrict__ esb,
                 const float* __restrict__ edw, const float* __restrict__ edb,
                 float* __restrict__ wsf) {
    int h = threadIdx.x;
    if (h >= Hd) return;
    const float* Wr = W + h * (3 * Hd);
    float a0 = 0.f, a1 = 0.f, bd = 0.f, cb = 0.f, q0 = 0.f, q1 = 0.f, qb = 0.f;
    for (int j = 0; j < Hd; ++j) {
        float we = Wr[j], wd = Wr[Hd + j], wh = Wr[2 * Hd + j];
        float e0 = esw[2 * j], e1 = esw[2 * j + 1];
        a0 = fmaf(we, e0, a0);
        a1 = fmaf(we, e1, a1);
        cb = fmaf(we, esb[j], cb);
        bd = fmaf(wd, edw[2 * j] + edw[2 * j + 1], bd);
        cb = fmaf(wd, edb[j], cb);
        q0 = fmaf(wh, e0, q0);
        q1 = fmaf(wh, e1, q1);
        qb = fmaf(wh, esb[j], qb);
    }
    wsf[0 * Hd + h] = a0;
    wsf[1 * Hd + h] = a1;
    wsf[2 * Hd + h] = bd;
    wsf[3 * Hd + h] = cb;
    wsf[4 * Hd + h] = q0;
    wsf[5 * Hd + h] = q1;
    wsf[6 * Hd + h] = qb;
}

// ---------------------------------------------------------------------------
// Kernel 2: per-batch 3-step attention refinement. One block per batch b.
// State per batch collapses to two scalars (px0, px1). Emits hy[b,:] as f16.
// ---------------------------------------------------------------------------
__global__ __launch_bounds__(256)
void attn_kernel(const float* __restrict__ stat, const float* __restrict__ dyn,
                 const float* __restrict__ esw, const float* __restrict__ esb,
                 const float* __restrict__ vvec, const float* __restrict__ wsf,
                 _Float16* __restrict__ hyh) {
    __shared__ float sx0[SEQ], sx1[SEQ], sy[SEQ];
    __shared__ float ca0[Hd], ca1[Hd], cbd[Hd], ccb[Hd], cq0[Hd], cq1[Hd], cqb[Hd];
    __shared__ float cv[Hd], ccc[Hd];
    __shared__ float slog[SEQ];
    __shared__ float red[256];
    __shared__ float sscal[4];   // [0] unused, [1]=px0, [2]=px1

    const int b = blockIdx.x, tid = threadIdx.x;
    const float* xs = stat + (size_t)b * 2 * SEQ;
    const float* yd = dyn + (size_t)b * 2 * SEQ + SEQ;   // dynamic channel 1
    for (int s = tid; s < SEQ; s += 256) {
        sx0[s] = xs[s];
        sx1[s] = xs[SEQ + s];
        sy[s]  = yd[s];
    }
    if (tid < Hd) {
        ca0[tid] = wsf[0 * Hd + tid];
        ca1[tid] = wsf[1 * Hd + tid];
        cbd[tid] = wsf[2 * Hd + tid];
        ccb[tid] = wsf[3 * Hd + tid];
        cq0[tid] = wsf[4 * Hd + tid];
        cq1[tid] = wsf[5 * Hd + tid];
        cqb[tid] = wsf[6 * Hd + tid];
        cv[tid]  = vvec[tid];
    }
    __syncthreads();

    for (int it = 0; it < 3; ++it) {
        // per-iteration constant term (includes W_h*hy folded through px0/px1)
        if (tid < Hd) {
            float cc = ccb[tid];
            if (it > 0)
                cc += fmaf(cq0[tid], sscal[1], fmaf(cq1[tid], sscal[2], cqb[tid]));
            ccc[tid] = cc;
        }
        __syncthreads();

        // logits: dominant work, H tanh per (b,s)
        for (int s = tid; s < SEQ; s += 256) {
            float x0 = sx0[s], x1 = sx1[s], yy = sy[s], acc = 0.f;
            #pragma unroll 4
            for (int h = 0; h < Hd; ++h) {
                float pre = fmaf(ca0[h], x0, fmaf(ca1[h], x1, fmaf(cbd[h], yy, ccc[h])));
                float th  = 1.0f - 2.0f / (__expf(2.0f * pre) + 1.0f);  // tanh
                acc = fmaf(cv[h], th, acc);
            }
            slog[s] = acc;
        }
        __syncthreads();

        // max-reduce (deterministic tree)
        float lmax = -3.4e38f;
        for (int s = tid; s < SEQ; s += 256) lmax = fmaxf(lmax, slog[s]);
        red[tid] = lmax;
        __syncthreads();
        for (int off = 128; off > 0; off >>= 1) {
            if (tid < off) red[tid] = fmaxf(red[tid], red[tid + off]);
            __syncthreads();
        }
        float m = red[0];
        __syncthreads();

        // sum exp, sum exp*x0, sum exp*x1
        float ls = 0.f, l0 = 0.f, l1 = 0.f;
        for (int s = tid; s < SEQ; s += 256) {
            float w = __expf(slog[s] - m);
            ls += w;
            l0 = fmaf(w, sx0[s], l0);
            l1 = fmaf(w, sx1[s], l1);
        }
        red[tid] = ls;
        __syncthreads();
        for (int off = 128; off > 0; off >>= 1) {
            if (tid < off) red[tid] += red[tid + off];
            __syncthreads();
        }
        float Z = red[0];
        __syncthreads();
        red[tid] = l0;
        __syncthreads();
        for (int off = 128; off > 0; off >>= 1) {
            if (tid < off) red[tid] += red[tid + off];
            __syncthreads();
        }
        float s0 = red[0];
        __syncthreads();
        red[tid] = l1;
        __syncthreads();
        for (int off = 128; off > 0; off >>= 1) {
            if (tid < off) red[tid] += red[tid + off];
            __syncthreads();
        }
        float s1 = red[0];
        __syncthreads();
        if (tid == 0) {
            sscal[1] = s0 / Z;
            sscal[2] = s1 / Z;
        }
        __syncthreads();
    }

    // hy[b,h] = esw[h,0]*px0 + esw[h,1]*px1 + esb[h]  (store f16 for WMMA head)
    if (tid < Hd) {
        float hy = fmaf(esw[2 * tid], sscal[1], fmaf(esw[2 * tid + 1], sscal[2], esb[tid]));
        hyh[(size_t)b * Hd + tid] = (_Float16)hy;
    }
}

// ---------------------------------------------------------------------------
// Kernel 3: head GEMM via v_wmma_f32_16x16x32_f16.
//   z = relu(hy @ dense_w^T + dense_b);  out = z @ lin_w^T + lin_b
// 16 blocks (M-tiles of 16 batches) x 8 waves (N-tiles of 16 cols), K=128.
// ---------------------------------------------------------------------------
__global__ __launch_bounds__(256)
void head_kernel(const _Float16* __restrict__ hyh,
                 const float* __restrict__ dense_w, const float* __restrict__ dense_b,
                 const float* __restrict__ lin_w,  const float* __restrict__ lin_b,
                 float* __restrict__ out) {
    __shared__ float zbuf[16 * 128];   // [row in M-tile][global N]
    const int tid  = threadIdx.x;
    const int lane = tid & 31;
    const int wave = tid >> 5;         // N-tile index 0..7
    const int r    = lane & 15;
    const int hsel = lane >> 4;
    const int brow = blockIdx.x * 16 + r;   // A row (batch)
    const int n    = wave * 16 + r;         // B column (dense output unit)

    v8f c = {};
    #pragma unroll
    for (int kk = 0; kk < 4; ++kk) {
        const int k0 = kk * 32;
        v16h a, bm;
        // A-matrix 16x32 f16 layout: lane holds row M=lane&15;
        // halves [0..15] = K = k0+16*hsel+j (contiguous, matches B)
        const _Float16* ap = hyh + (size_t)brow * Hd + k0 + 16 * hsel;
        #pragma unroll
        for (int j = 0; j < 8; ++j) {
            a[j]     = ap[j];
            a[8 + j] = ap[8 + j];
        }
        // B-matrix 32x16: lane holds col N=lane&15; halves m = B[k0+16*hsel+m, N]
        // B[k,n] = dense_w[n,k]  (row-major (128,128)) -> 16 contiguous floats
        const float* bp = dense_w + (size_t)n * Hd + k0 + 16 * hsel;
        #pragma unroll
        for (int m = 0; m < 16; ++m) bm[m] = (_Float16)bp[m];
        c = __builtin_amdgcn_wmma_f32_16x16x32_f16(false, a, false, bm,
                                                   (short)0, c, false, false);
    }

    const float bias = dense_b[n];
    const float lw   = lin_w[n];
    #pragma unroll
    for (int g = 0; g < 8; ++g) {
        float z = c[g] + bias;           // D[M,N], M = g + 8*hsel
        z = z > 0.f ? z : 0.f;           // relu
        const int M = g + 8 * hsel;
        zbuf[M * 128 + wave * 16 + r] = z * lw;
    }
    __syncthreads();
    if (tid < 16) {                      // deterministic per-row reduction over N=128
        float s = 0.f;
        for (int j = 0; j < 128; ++j) s += zbuf[tid * 128 + j];
        out[blockIdx.x * 16 + tid] = s + lin_b[0];
    }
}

// ---------------------------------------------------------------------------
extern "C" void kernel_launch(void* const* d_in, const int* in_sizes, int n_in,
                              void* d_out, int out_size, void* d_ws, size_t ws_size,
                              hipStream_t stream) {
    (void)in_sizes; (void)n_in; (void)out_size; (void)ws_size;
    const float* stat    = (const float*)d_in[0];
    const float* dyn     = (const float*)d_in[1];
    const float* enc_s_w = (const float*)d_in[2];
    const float* enc_s_b = (const float*)d_in[3];
    const float* enc_d_w = (const float*)d_in[4];
    const float* enc_d_b = (const float*)d_in[5];
    const float* vvec    = (const float*)d_in[6];
    const float* W       = (const float*)d_in[7];
    const float* dense_w = (const float*)d_in[8];
    const float* dense_b = (const float*)d_in[9];
    const float* lin_w   = (const float*)d_in[10];
    const float* lin_b   = (const float*)d_in[11];
    float* out = (float*)d_out;

    float* wsf = (float*)d_ws;                                    // 7*128 floats
    _Float16* hyh = (_Float16*)((char*)d_ws + 4096);              // 256*128 f16

    fold_kernel<<<1, 128, 0, stream>>>(W, enc_s_w, enc_s_b, enc_d_w, enc_d_b, wsf);
    attn_kernel<<<BATCH, 256, 0, stream>>>(stat, dyn, enc_s_w, enc_s_b, vvec, wsf, hyh);
    head_kernel<<<BATCH / 16, 256, 0, stream>>>(hyh, dense_w, dense_b, lin_w, lin_b, out);
}
